// ORCDF_Extractor_13572096655715
// MI455X (gfx1250) — matrix operand order repaired and checked by reading, not compiled, not run on hardware
//
#include <hip/hip_runtime.h>
#include <math.h>

// ---------------------------------------------------------------------------
// ORCDF extractor pipeline for MI455X (gfx1250, wave32, WMMA bf16).
// ---------------------------------------------------------------------------

#define S_NUM_C   10000
#define E_NUM_C   6000
#define K_NUM_C   400
#define N_TOTAL_C 16400
#define TEXT_DIM_C 768
#define LATENT_C  128
#define BATCH_C   4096
#define SSL_TEMP_C 0.5f
#define SSL_WEIGHT_C 0.01f
#define KEEP_PROB_C 0.9f

typedef __attribute__((ext_vector_type(16))) __bf16 v16bf;
typedef __attribute__((ext_vector_type(8)))  float  v8f;

union BFU { uint4 u4[2]; v16bf v; };

__device__ __forceinline__ unsigned short f2bf(float x) {
  unsigned int u = __float_as_uint(x);
  return (unsigned short)((u + 0x7FFFu + ((u >> 16) & 1u)) >> 16);
}

// A-matrix 16x32 bf16 tile (contiguous k order, not the ISA table order):
//  lanes 0-15: K = kb+0..15; lanes 16-31: K = kb+16..31 (matches B)
__device__ __forceinline__ v16bf ldtileA(const unsigned short* base, int ldk,
                                         int row, int kb, int hf) {
  const unsigned short* p = base + (size_t)row * (size_t)ldk + kb + hf * 16;
  BFU t;
  t.u4[0] = *(const uint4*)(const void*)p;
  t.u4[1] = *(const uint4*)(const void*)(p + 8);
  return t.v;
}

// B-matrix 32x16 bf16 tile (contiguous k order, same as ldtileA):
//  lane = column N; lanes 0-15 hold K=kb+0..15, lanes 16-31 hold K=kb+16..31
// Source is row-major W(N,K): 16 contiguous K values at offset kb + hf*16.
__device__ __forceinline__ v16bf ldtileB(const unsigned short* base, int ldk,
                                         int row, int kb, int hf) {
  const unsigned short* p = base + (size_t)row * (size_t)ldk + kb + hf * 16;
  BFU t;
  t.u4[0] = *(const uint4*)(const void*)p;
  t.u4[1] = *(const uint4*)(const void*)(p + 8);
  return t.v;
}

__device__ __forceinline__ v16bf ldtileB_g(const unsigned short* base, int ldk,
                                           int row, int kb, int hf, int nrows) {
  BFU t;
  if (row < nrows) {
    const unsigned short* p = base + (size_t)row * (size_t)ldk + kb + hf * 16;
    t.u4[0] = *(const uint4*)(const void*)p;
    t.u4[1] = *(const uint4*)(const void*)(p + 8);
  } else {
    t.u4[0] = make_uint4(0u, 0u, 0u, 0u);
    t.u4[1] = make_uint4(0u, 0u, 0u, 0u);
  }
  return t.v;
}

#define WMMA_BF16(A, B, C) \
  __builtin_amdgcn_wmma_f32_16x16x32_bf16(false, (A), false, (B), (short)0, (C), false, false)

__device__ __forceinline__ float redmax16(float x) {
  x = fmaxf(x, __shfl_xor(x, 1, 32));
  x = fmaxf(x, __shfl_xor(x, 2, 32));
  x = fmaxf(x, __shfl_xor(x, 4, 32));
  x = fmaxf(x, __shfl_xor(x, 8, 32));
  return x;
}
__device__ __forceinline__ float redsum16(float x) {
  x += __shfl_xor(x, 1, 32);
  x += __shfl_xor(x, 2, 32);
  x += __shfl_xor(x, 4, 32);
  x += __shfl_xor(x, 8, 32);
  return x;
}

// ---------------------------------------------------------------------------
// Elementwise / utility kernels
// ---------------------------------------------------------------------------

__global__ void k_cvt_bf16(const float* __restrict__ in,
                           unsigned short* __restrict__ out, int n) {
  int i = blockIdx.x * 256 + threadIdx.x;
  if (i < n) out[i] = f2bf(in[i]);
}

__global__ void k_pack_cat(const float* __restrict__ re,
                           const float* __restrict__ we,
                           unsigned short* __restrict__ cat, int rows) {
  int i = blockIdx.x * 256 + threadIdx.x;
  if (i < rows * 256) {
    int r = i >> 8, d = i & 255;
    float v = (d < 128) ? re[r * 128 + d] : we[r * 128 + (d - 128)];
    cat[i] = f2bf(v);
  }
}

__global__ void k_axpy(float* __restrict__ acc, const float* __restrict__ x, int n) {
  int i = blockIdx.x * 256 + threadIdx.x;
  if (i < n) acc[i] += x[i];
}

__global__ void k_scale(float* __restrict__ x, float s, int n) {
  int i = blockIdx.x * 256 + threadIdx.x;
  if (i < n) x[i] *= s;
}

__global__ void k_copy(const float* __restrict__ src, float* __restrict__ dst, int n) {
  int i = blockIdx.x * 256 + threadIdx.x;
  if (i < n) dst[i] = src[i];
}

__global__ void k_gather_rows(const float* __restrict__ src,
                              const int* __restrict__ ids, int rowoff,
                              float* __restrict__ dst, int n, int dim) {
  int i = blockIdx.x * 256 + threadIdx.x;
  if (i < n * dim) {
    int r = i / dim, d = i - r * dim;
    dst[i] = src[(size_t)(ids[r] + rowoff) * dim + d];
  }
}

__global__ void k_gather_scalar(const float* __restrict__ src,
                                const int* __restrict__ ids,
                                float* __restrict__ dst, int n) {
  int i = blockIdx.x * 256 + threadIdx.x;
  if (i < n) dst[i] = src[ids[i]];
}

// ---------------------------------------------------------------------------
// SPMM with fused edge dropout (deterministic per-edge hash RNG)
// ---------------------------------------------------------------------------

__global__ void k_spmm_drop(const int* __restrict__ rows,
                            const int* __restrict__ cols,
                            const float* __restrict__ vals,
                            const float* __restrict__ x,
                            float* __restrict__ y,
                            int nnz, unsigned int seed) {
  long long idx = (long long)blockIdx.x * 256 + threadIdx.x;
  int e = (int)(idx >> 7);
  int d = (int)(idx & 127);
  if (e < nnz) {
    unsigned int h = ((unsigned int)e * 2654435761u) ^ (seed * 0x9E3779B9u);
    h ^= h >> 16; h *= 0x85EBCA6Bu; h ^= h >> 13; h *= 0xC2B2AE35u; h ^= h >> 16;
    float u = (float)(h >> 8) * (1.0f / 16777216.0f);
    if (u < KEEP_PROB_C) {
      float v = vals[e] * (1.0f / KEEP_PROB_C);
      atomicAdd(&y[(size_t)rows[e] * 128 + d], v * x[(size_t)cols[e] * 128 + d]);
    }
  }
}

// ---------------------------------------------------------------------------
// WMMA GEMM (fast path): C(M,N) = A(M,K)bf16 @ W(N,K)bf16^T + bias (+PReLU).
// Requires N % 64 == 0, M % 16 == 0, K % 32 == 0 (true for every layer except
// the disc head). One wave computes a 16x64 stripe: 4 accumulators, A tile
// reused 4x, no exec-mask manipulation or branches in the inner loop.
// Prefetch is unconditional: speculative prefetch past the end of a buffer is
// architecturally dropped, so no guard (and no per-iteration s_cmp/s_cbranch).
// ---------------------------------------------------------------------------

__global__ void k_gemm_wmma_fast(const unsigned short* __restrict__ A,
                                 const unsigned short* __restrict__ W,
                                 const float* __restrict__ bias,
                                 const float* __restrict__ alpha_p,
                                 float* __restrict__ Cf,
                                 unsigned short* __restrict__ Cb,
                                 int M, int N, int K, int act) {
  int gwave = (blockIdx.x * blockDim.x + threadIdx.x) >> 5;
  int lane  = threadIdx.x & 31;
  int nt = N >> 6;
  int mt = gwave / nt;
  int ntile = gwave - mt * nt;
  if (mt * 16 >= M) return;
  int l16 = lane & 15, hf = lane >> 4;
  int rowA = mt * 16 + l16;
  int nbase = ntile << 6;

  const unsigned short* pa  = A + (size_t)rowA * (size_t)K;
  const unsigned short* pb0 = W + (size_t)(nbase + l16) * (size_t)K;

  v8f acc[4];
#pragma unroll
  for (int t = 0; t < 4; t++) acc[t] = (v8f){0.f,0.f,0.f,0.f,0.f,0.f,0.f,0.f};

  for (int k = 0; k < K; k += 32) {
    // stream-ahead hint into GL2 for the next K slab (unguarded: speculative)
    __builtin_prefetch(pa + k + 32, 0, 0);
    __builtin_prefetch(pb0 + k + 32, 0, 0);
    v16bf a  = ldtileA(A, K, rowA, k, hf);
    v16bf b0 = ldtileB(W, K, nbase + l16,      k, hf);
    v16bf b1 = ldtileB(W, K, nbase + 16 + l16, k, hf);
    v16bf b2 = ldtileB(W, K, nbase + 32 + l16, k, hf);
    v16bf b3 = ldtileB(W, K, nbase + 48 + l16, k, hf);
    acc[0] = WMMA_BF16(a, b0, acc[0]);
    acc[1] = WMMA_BF16(a, b1, acc[1]);
    acc[2] = WMMA_BF16(a, b2, acc[2]);
    acc[3] = WMMA_BF16(a, b3, acc[3]);
  }

  float alpha = (act != 0 && alpha_p) ? alpha_p[0] : 0.0f;
#pragma unroll
  for (int t = 0; t < 4; t++) {
    int col = nbase + t * 16 + l16;
    float bv = bias ? bias[col] : 0.0f;
#pragma unroll
    for (int r = 0; r < 8; r++) {
      float v = acc[t][r] + bv;
      if (act) v = (v >= 0.0f) ? v : alpha * v;
      size_t o = (size_t)(mt * 16 + r + 8 * hf) * (size_t)N + col;
      if (Cf) Cf[o] = v;
      else    Cb[o] = f2bf(v);
    }
  }
}

// ---------------------------------------------------------------------------
// WMMA GEMM (guarded path): handles ragged N (disc head, N == 1).
// ---------------------------------------------------------------------------

__global__ void k_gemm_wmma_guard(const unsigned short* __restrict__ A,
                                  const unsigned short* __restrict__ W,
                                  const float* __restrict__ bias,
                                  float* __restrict__ Cf,
                                  int M, int N, int K) {
  int gwave = (blockIdx.x * blockDim.x + threadIdx.x) >> 5;
  int lane  = threadIdx.x & 31;
  int nt = (N + 15) >> 4;
  int mt = gwave / nt;
  int ntile = gwave - mt * nt;
  if (mt * 16 >= M) return;
  int l16 = lane & 15, hf = lane >> 4;
  int rowA = mt * 16 + l16;
  int nbase = ntile << 4;

  v8f acc = (v8f){0.f,0.f,0.f,0.f,0.f,0.f,0.f,0.f};
  for (int k = 0; k < K; k += 32) {
    v16bf a = ldtileA(A, K, rowA, k, hf);
    v16bf b = ldtileB_g(W, K, nbase + l16, k, hf, N);
    acc = WMMA_BF16(a, b, acc);
  }
  int col = nbase + l16;
  if (col < N) {
    float bv = bias ? bias[col] : 0.0f;
#pragma unroll
    for (int r = 0; r < 8; r++) {
      size_t o = (size_t)(mt * 16 + r + 8 * hf) * (size_t)N + col;
      Cf[o] = acc[r] + bv;
    }
  }
}

// ---------------------------------------------------------------------------
// Fused InfoNCE: per-wave 16-row block of V1, streams V2 in 16-col tiles,
// WMMA computes 16x16 logits, online logsumexp kept in registers.
// loss += sc * sum_i (logsumexp_j(l_ij) - l_ii),  l = (V1 V2^T)/T
// ---------------------------------------------------------------------------

__global__ void k_nce_wmma(const unsigned short* __restrict__ V1,
                           const unsigned short* __restrict__ V2,
                           int R, float sc, float* __restrict__ loss) {
  int gwave = (blockIdx.x * blockDim.x + threadIdx.x) >> 5;
  int lane  = threadIdx.x & 31;
  int mbase = gwave * 16;
  if (mbase >= R) return;
  int l16 = lane & 15, hf = lane >> 4;
  int rowA = mbase + l16;

  v16bf a0 = ldtileA(V1, 128, rowA,  0, hf);
  v16bf a1 = ldtileA(V1, 128, rowA, 32, hf);
  v16bf a2 = ldtileA(V1, 128, rowA, 64, hf);
  v16bf a3 = ldtileA(V1, 128, rowA, 96, hf);

  float m[8], s[8], dg[8];
#pragma unroll
  for (int r = 0; r < 8; r++) { m[r] = -3.0e38f; s[r] = 0.0f; dg[r] = 0.0f; }

  for (int nb = 0; nb < R; nb += 16) {
    int rowB = nb + l16;
    v16bf b0 = ldtileB(V2, 128, rowB,  0, hf);
    v16bf b1 = ldtileB(V2, 128, rowB, 32, hf);
    v16bf b2 = ldtileB(V2, 128, rowB, 64, hf);
    v16bf b3 = ldtileB(V2, 128, rowB, 96, hf);
    v8f acc = (v8f){0.f,0.f,0.f,0.f,0.f,0.f,0.f,0.f};
    acc = WMMA_BF16(a0, b0, acc);
    acc = WMMA_BF16(a1, b1, acc);
    acc = WMMA_BF16(a2, b2, acc);
    acc = WMMA_BF16(a3, b3, acc);
#pragma unroll
    for (int r = 0; r < 8; r++) {
      float x = acc[r] * (1.0f / SSL_TEMP_C);
      int grow = mbase + r + 8 * hf;
      if (nb + l16 == grow) dg[r] = x;
      float tm = redmax16(x);
      float ex = redsum16(__expf(x - tm));
      float nm = fmaxf(m[r], tm);
      s[r] = s[r] * __expf(m[r] - nm) + ex * __expf(tm - nm);
      m[r] = nm;
    }
  }

#pragma unroll
  for (int r = 0; r < 8; r++) {
    float d = redsum16(dg[r]);  // exactly one lane/tile saw the diagonal
    if (l16 == 0) {
      atomicAdd(loss, sc * ((m[r] + __logf(s[r])) - d));
    }
  }
}

// ---------------------------------------------------------------------------
// Host orchestration
// ---------------------------------------------------------------------------

extern "C" void kernel_launch(void* const* d_in, const int* in_sizes, int n_in,
                              void* d_out, int out_size, void* d_ws, size_t ws_size,
                              hipStream_t stream) {
  (void)n_in; (void)out_size; (void)ws_size;

  const float* emb_s = (const float*)d_in[0];
  const float* emb_e = (const float*)d_in[1];
  const float* emb_k = (const float*)d_in[2];
  const float* mlp[4][8];
  for (int mI = 0; mI < 4; mI++)
    for (int j = 0; j < 8; j++)
      mlp[mI][j] = (const float*)d_in[3 + 8 * mI + j];
  const float* concat_W = (const float*)d_in[35];
  const float* concat_b = (const float*)d_in[36];
  const int*   gr_rows[4] = {(const int*)d_in[37], (const int*)d_in[40],
                             (const int*)d_in[43], (const int*)d_in[46]};
  const int*   gr_cols[4] = {(const int*)d_in[38], (const int*)d_in[41],
                             (const int*)d_in[44], (const int*)d_in[47]};
  const float* gr_vals[4] = {(const float*)d_in[39], (const float*)d_in[42],
                             (const float*)d_in[45], (const float*)d_in[48]};
  int nnz[4] = {in_sizes[39], in_sizes[42], in_sizes[45], in_sizes[48]};
  const int* student_id  = (const int*)d_in[49];
  const int* exercise_id = (const int*)d_in[50];

  char* wp = (char*)d_ws;
  auto carve = [&](size_t bytes) -> void* {
    void* r = wp;
    wp += (bytes + 255) & ~(size_t)255;
    return r;
  };

  const int dout_mlp[4] = {128, 128, 128, 1};

  unsigned short* sbf = (unsigned short*)carve((size_t)S_NUM_C * TEXT_DIM_C * 2);
  unsigned short* ebf = (unsigned short*)carve((size_t)E_NUM_C * TEXT_DIM_C * 2);
  unsigned short* kbf = (unsigned short*)carve((size_t)K_NUM_C * TEXT_DIM_C * 2);
  unsigned short* w1bf[4]; unsigned short* w2bf[4]; unsigned short* w3bf[4];
  for (int mI = 0; mI < 4; mI++) {
    w1bf[mI] = (unsigned short*)carve((size_t)512 * 768 * 2);
    w2bf[mI] = (unsigned short*)carve((size_t)256 * 512 * 2);
    w3bf[mI] = (unsigned short*)carve((size_t)dout_mlp[mI] * 256 * 2);
  }
  unsigned short* cwbf  = (unsigned short*)carve((size_t)128 * 256 * 2);
  unsigned short* h1bf  = (unsigned short*)carve((size_t)S_NUM_C * 512 * 2);
  unsigned short* h2bf  = (unsigned short*)carve((size_t)S_NUM_C * 256 * 2);
  float* all_emb  = (float*)carve((size_t)N_TOTAL_C * LATENT_C * 4);
  float* disc_out = (float*)carve((size_t)E_NUM_C * 4);
  float* gbuf[4];
  for (int b = 0; b < 4; b++) gbuf[b] = (float*)carve((size_t)N_TOTAL_C * LATENT_C * 4);
  unsigned short* catbf = (unsigned short*)carve((size_t)N_TOTAL_C * 256 * 2);
  float* layer_out = (float*)carve((size_t)N_TOTAL_C * LATENT_C * 4);
  float* accA = (float*)carve((size_t)N_TOTAL_C * LATENT_C * 4);
  float* accB = (float*)carve((size_t)N_TOTAL_C * LATENT_C * 4);
  unsigned short* nce1 = (unsigned short*)carve((size_t)S_NUM_C * 128 * 2);
  unsigned short* nce2 = (unsigned short*)carve((size_t)S_NUM_C * 128 * 2);

  auto cvt = [&](const float* in, unsigned short* out, int n) {
    k_cvt_bf16<<<(n + 255) / 256, 256, 0, stream>>>(in, out, n);
  };
  auto gemm = [&](const unsigned short* A, const unsigned short* W,
                  const float* bias, const float* alpha, float* Cf,
                  unsigned short* Cb, int M, int N, int K, int act) {
    if ((N & 63) == 0) {
      int waves = (M / 16) * (N / 64);
      int blocks = (waves * 32 + 255) / 256;
      k_gemm_wmma_fast<<<blocks, 256, 0, stream>>>(A, W, bias, alpha, Cf, Cb,
                                                   M, N, K, act);
    } else {
      int waves = (M / 16) * ((N + 15) / 16);
      int blocks = (waves * 32 + 255) / 256;
      k_gemm_wmma_guard<<<blocks, 256, 0, stream>>>(A, W, bias, Cf, M, N, K);
    }
  };
  auto spmm = [&](int g, const float* x, float* y, unsigned int seed) {
    long long tot = (long long)nnz[g] * 128;
    int blocks = (int)((tot + 255) / 256);
    k_spmm_drop<<<blocks, 256, 0, stream>>>(gr_rows[g], gr_cols[g], gr_vals[g],
                                            x, y, nnz[g], seed);
  };

  // ---- bf16 conversions of inputs & weights ----
  cvt(emb_s, sbf, S_NUM_C * TEXT_DIM_C);
  cvt(emb_e, ebf, E_NUM_C * TEXT_DIM_C);
  cvt(emb_k, kbf, K_NUM_C * TEXT_DIM_C);
  for (int mI = 0; mI < 4; mI++) {
    cvt(mlp[mI][0], w1bf[mI], 512 * 768);
    cvt(mlp[mI][3], w2bf[mI], 256 * 512);
    cvt(mlp[mI][6], w3bf[mI], dout_mlp[mI] * 256);
  }
  cvt(concat_W, cwbf, 128 * 256);

  // ---- entity MLPs (WMMA GEMM chain, PReLU fused, bf16 intermediates) ----
  auto run_mlp = [&](const unsigned short* inbf, int M, int mI, float* out_f32, int dout) {
    gemm(inbf, w1bf[mI], mlp[mI][1], mlp[mI][2], nullptr, h1bf, M, 512, 768, 1);
    gemm(h1bf, w2bf[mI], mlp[mI][4], mlp[mI][5], nullptr, h2bf, M, 256, 512, 1);
    gemm(h2bf, w3bf[mI], mlp[mI][7], nullptr, out_f32, nullptr, M, dout, 256, 0);
  };
  run_mlp(sbf, S_NUM_C, 0, all_emb, 128);
  run_mlp(ebf, E_NUM_C, 1, all_emb + (size_t)S_NUM_C * 128, 128);
  run_mlp(kbf, K_NUM_C, 2, all_emb + (size_t)(S_NUM_C + E_NUM_C) * 128, 128);
  run_mlp(ebf, E_NUM_C, 3, disc_out, 1);

  // ---- common_forward x2 (3 GCN layers each) ----
  size_t nbytes = (size_t)N_TOTAL_C * LATENT_C * sizeof(float);
  auto run_forward = [&](int gr, int gw, float* acc, unsigned int seedbase) {
    float* re = gbuf[0]; float* we = gbuf[1];
    float* t1 = gbuf[2]; float* t2 = gbuf[3];
    hipMemcpyAsync(re, all_emb, nbytes, hipMemcpyDeviceToDevice, stream);
    hipMemcpyAsync(we, all_emb, nbytes, hipMemcpyDeviceToDevice, stream);
    hipMemcpyAsync(acc, all_emb, nbytes, hipMemcpyDeviceToDevice, stream);
    for (int l = 0; l < 3; l++) {
      hipMemsetAsync(t1, 0, nbytes, stream);
      hipMemsetAsync(t2, 0, nbytes, stream);
      spmm(gr, re, t1, seedbase + 2 * l);
      spmm(gw, we, t2, seedbase + 2 * l + 1);
      float* tmp;
      tmp = re; re = t1; t1 = tmp;
      tmp = we; we = t2; t2 = tmp;
      int npk = N_TOTAL_C * 256;
      k_pack_cat<<<(npk + 255) / 256, 256, 0, stream>>>(re, we, catbf, N_TOTAL_C);
      gemm(catbf, cwbf, concat_b, nullptr, layer_out, nullptr, N_TOTAL_C, 128, 256, 0);
      int nel = N_TOTAL_C * LATENT_C;
      k_axpy<<<(nel + 255) / 256, 256, 0, stream>>>(acc, layer_out, nel);
    }
    int nel = N_TOTAL_C * LATENT_C;
    k_scale<<<(nel + 255) / 256, 256, 0, stream>>>(acc, 0.25f, nel);
  };
  run_forward(0, 1, accA, 0x11111111u);   // (right, wrong)      -> stu_f/exer_f/know_f
  run_forward(2, 3, accB, 0x22222222u);   // (right_f, wrong_f)  -> stu_ff/exer_ff

  // ---- InfoNCE (fused WMMA + online logsumexp) ----
  float* out_f = (float*)d_out;
  float* lossp = out_f + (size_t)2 * BATCH_C * 128 + BATCH_C + (size_t)K_NUM_C * 128;
  hipMemsetAsync(lossp, 0, sizeof(float), stream);

  cvt(accA, nce1, S_NUM_C * 128);
  cvt(accB, nce2, S_NUM_C * 128);
  {
    int waves = S_NUM_C / 16;
    int blocks = (waves * 32 + 255) / 256;
    k_nce_wmma<<<blocks, 256, 0, stream>>>(nce1, nce2, S_NUM_C,
                                           SSL_WEIGHT_C / (float)S_NUM_C, lossp);
  }
  cvt(accA + (size_t)S_NUM_C * 128, nce1, E_NUM_C * 128);
  cvt(accB + (size_t)S_NUM_C * 128, nce2, E_NUM_C * 128);
  {
    int waves = E_NUM_C / 16;
    int blocks = (waves * 32 + 255) / 256;
    k_nce_wmma<<<blocks, 256, 0, stream>>>(nce1, nce2, E_NUM_C,
                                           SSL_WEIGHT_C / (float)E_NUM_C, lossp);
  }

  // ---- output gathers ----
  {
    int n = BATCH_C * 128;
    k_gather_rows<<<(n + 255) / 256, 256, 0, stream>>>(accA, student_id, 0,
                                                       out_f, BATCH_C, 128);
    k_gather_rows<<<(n + 255) / 256, 256, 0, stream>>>(accA, exercise_id, S_NUM_C,
                                                       out_f + (size_t)BATCH_C * 128,
                                                       BATCH_C, 128);
    k_gather_scalar<<<(BATCH_C + 255) / 256, 256, 0, stream>>>(
        disc_out, exercise_id, out_f + (size_t)2 * BATCH_C * 128, BATCH_C);
    int nk = K_NUM_C * 128;
    k_copy<<<(nk + 255) / 256, 256, 0, stream>>>(
        accA + (size_t)(S_NUM_C + E_NUM_C) * 128,
        out_f + (size_t)2 * BATCH_C * 128 + BATCH_C, nk);
  }
}
